// MaxKSAGEConv_62388694942254
// MI455X (gfx1250) — matrix operand order repaired and checked by reading, not compiled, not run on hardware
//
#include <hip/hip_runtime.h>
#include <stdint.h>

#define IN_FEATS 128
#define OUT_FEATS 256
#define KTOP 32
#define LDS_STRIDE (IN_FEATS + 1)
#define N_STEPS 8              // K = 256 / 32
#define N_NTILES 16            // 256 / 16

typedef __attribute__((ext_vector_type(16))) __bf16    v16bf;
typedef __attribute__((ext_vector_type(8)))  float     v8f;
typedef __attribute__((ext_vector_type(8)))  unsigned  v8u;

// ---- bf16 split helpers (native converts) --------------------------------
static __device__ __forceinline__ void split_pack(float x0, float x1,
                                                  unsigned& hi_pair, unsigned& lo_pair) {
    __bf16 h0 = (__bf16)x0;
    __bf16 h1 = (__bf16)x1;
    __bf16 l0 = (__bf16)(x0 - (float)h0);
    __bf16 l1 = (__bf16)(x1 - (float)h1);
    unsigned uh0 = __builtin_bit_cast(unsigned short, h0);
    unsigned uh1 = __builtin_bit_cast(unsigned short, h1);
    unsigned ul0 = __builtin_bit_cast(unsigned short, l0);
    unsigned ul1 = __builtin_bit_cast(unsigned short, l1);
    hi_pair = uh0 | (uh1 << 16);
    lo_pair = ul0 | (ul1 << 16);
}

// ---- kernel 1: COO(sorted rows) -> indptr --------------------------------
__global__ void build_indptr_kernel(const int* __restrict__ row,
                                    int* __restrict__ indptr,
                                    int n_edges, int n_nodes) {
    int e = blockIdx.x * blockDim.x + threadIdx.x;
    if (e >= n_edges) return;
    int r     = row[e];
    int rprev = (e == 0) ? -1 : row[e - 1];
    for (int q = rprev + 1; q <= r; ++q) indptr[q] = e;
    if (e == n_edges - 1) {
        for (int q = r + 1; q <= n_nodes; ++q) indptr[q] = n_edges;
    }
}

// ---- kernel 2: pack W' = [W_self ; W_neigh] into WMMA B-fragment layout --
// B frag (32x16 bf16, wave32): lane = hl*16 + l16; col n = ntile*16 + l16;
// VGPR r holds K pair (hl*16 + 2r, +1). One uint32 per (step, ntile, lane, r).
__global__ void pack_W_kernel(const float* __restrict__ W_self,
                              const float* __restrict__ W_neigh,
                              unsigned* __restrict__ bhi,
                              unsigned* __restrict__ blo) {
    int t = blockIdx.x * blockDim.x + threadIdx.x;   // 0 .. 8*16*32-1
    if (t >= N_STEPS * N_NTILES * 32) return;
    int lane = t & 31;
    int nt   = (t >> 5) & 15;
    int step = t >> 9;
    int hl   = lane >> 4;
    int l16  = lane & 15;
    int n    = nt * 16 + l16;
    int k0   = step * 32;
    const float* Wk = (k0 < IN_FEATS) ? (W_self + (size_t)k0 * OUT_FEATS)
                                      : (W_neigh + (size_t)(k0 - IN_FEATS) * OUT_FEATS);
    #pragma unroll
    for (int r = 0; r < 8; ++r) {
        int kk = hl * 16 + 2 * r;
        float x0 = Wk[(size_t)kk * OUT_FEATS + n];
        float x1 = Wk[(size_t)(kk + 1) * OUT_FEATS + n];
        unsigned hp, lp; split_pack(x0, x1, hp, lp);
        bhi[(size_t)t * 8 + r] = hp;
        blo[(size_t)t * 8 + r] = lp;
    }
}

// ---- kernel 3: fused aggregate(topk) + normalize + dual-GEMM (WMMA) ------
__global__ __launch_bounds__(256)
void maxk_sage_fused_kernel(const float* __restrict__ feat,
                            const float* __restrict__ topk_val,
                            const int*   __restrict__ topk_idx,
                            const int*   __restrict__ col,
                            const int*   __restrict__ indptr,
                            const float* __restrict__ degrees,
                            const unsigned* __restrict__ bhi,
                            const unsigned* __restrict__ blo,
                            const float* __restrict__ bias,
                            float* __restrict__ out,
                            int n_nodes) {
    __shared__ float agg_s[16][LDS_STRIDE];
    // A fragments (16x32 bf16 per step), hi/lo: [step][lane][r] as uint32
    __shared__ __align__(16) unsigned afrag_hi[N_STEPS * 32 * 8];
    __shared__ __align__(16) unsigned afrag_lo[N_STEPS * 32 * 8];

    const int m0   = blockIdx.x * 16;
    const int tid  = threadIdx.x;
    const int wave = tid >> 5;     // 0..7
    const int lane = tid & 31;     // wave32 lane
    const int hl   = lane >> 4;
    const int l16  = lane & 15;

    // -- zero the LDS accumulator
    for (int i = tid; i < 16 * LDS_STRIDE; i += 256)
        (&agg_s[0][0])[i] = 0.0f;
    __syncthreads();

    // -- aggregation: wave w handles local rows {w, w+8}; lane j = topk slot j
    for (int rr = 0; rr < 2; ++rr) {
        int r_local = wave + rr * 8;
        int r = m0 + r_local;
        if (r < n_nodes) {
            int e0 = indptr[r], e1 = indptr[r + 1];
            for (int e = e0; e < e1; ++e) {
                int   c = col[e];
                float v = topk_val[(size_t)c * KTOP + lane];   // coalesced 128B
                int   f = topk_idx[(size_t)c * KTOP + lane];
                atomicAdd(&agg_s[r_local][f], v);              // ds_add_f32
            }
        }
    }
    __syncthreads();

    // -- mean normalization
    for (int i = tid; i < 16 * IN_FEATS; i += 256) {
        int r = i >> 7, f = i & 127;
        int rg = m0 + r;
        float d = (rg < n_nodes) ? degrees[rg] : 1.0f;
        agg_s[r][f] *= (1.0f / d);
    }
    __syncthreads();

    // -- cooperative A-tile pack: [feat | agg] -> bf16 hi/lo fragment layout
    // A frag: lane = hl*16 + l16; row m = l16; VGPR r holds K pair at
// klocal = 2r + hl*16, kglobal = step*32 + klocal.
    for (int p = tid; p < N_STEPS * 32 * 8; p += 256) {
        int step  = p >> 8;
        int ln    = (p >> 3) & 31;
        int r     = p & 7;
        int hlp   = ln >> 4;
        int l16p  = ln & 15;
        int kl    = 2 * r + hlp * 16;
        int kg    = step * 32 + kl;
        int arow  = (m0 + l16p < n_nodes) ? (m0 + l16p) : (n_nodes - 1);
        float x0, x1;
        if (kg < IN_FEATS) {
            const float* f = feat + (size_t)arow * IN_FEATS;
            x0 = f[kg]; x1 = f[kg + 1];
        } else {
            x0 = agg_s[l16p][kg - IN_FEATS];
            x1 = agg_s[l16p][kg - IN_FEATS + 1];
        }
        unsigned hp, lp; split_pack(x0, x1, hp, lp);
        afrag_hi[p] = hp;
        afrag_lo[p] = lp;
    }
    __syncthreads();

    // -- GEMM: acc += Ah*Bh + Ah*Bl + Al*Bh  (bf16 hi/lo split, near-fp32)
    v8f acc0 = {}; v8f acc1 = {};

    #pragma unroll
    for (int step = 0; step < N_STEPS; ++step) {
        v8u ah_u = *(const v8u*)&afrag_hi[(step * 32 + lane) * 8];  // ds_load_b128 x2
        v8u al_u = *(const v8u*)&afrag_lo[(step * 32 + lane) * 8];
        v16bf ah = __builtin_bit_cast(v16bf, ah_u);
        v16bf al = __builtin_bit_cast(v16bf, al_u);

        #pragma unroll
        for (int t = 0; t < 2; ++t) {
            size_t boff = ((size_t)(step * N_NTILES + (wave + t * 8)) * 32 + lane) * 8;
            v8u bh_u = *(const v8u*)&bhi[boff];                     // global_load_b128 x2
            v8u bl_u = *(const v8u*)&blo[boff];
            v16bf bh = __builtin_bit_cast(v16bf, bh_u);
            v16bf bl = __builtin_bit_cast(v16bf, bl_u);

            v8f a = (t == 0) ? acc0 : acc1;
            a = __builtin_amdgcn_wmma_f32_16x16x32_bf16(false, ah, false, bl,
                                                        (short)0, a, false, false);
            a = __builtin_amdgcn_wmma_f32_16x16x32_bf16(false, al, false, bh,
                                                        (short)0, a, false, false);
            a = __builtin_amdgcn_wmma_f32_16x16x32_bf16(false, ah, false, bh,
                                                        (short)0, a, false, false);
            if (t == 0) acc0 = a; else acc1 = a;
        }
    }

    // -- bias + store (C/D layout: VGPR r -> row r + 8*hl, col l16)
    #pragma unroll
    for (int t = 0; t < 2; ++t) {
        int n = (wave + t * 8) * 16 + l16;
        float b = bias[n];
        v8f a = (t == 0) ? acc0 : acc1;
        #pragma unroll
        for (int r = 0; r < 8; ++r) {
            int m = m0 + r + hl * 8;
            if (m < n_nodes)
                out[(size_t)m * OUT_FEATS + n] = a[r] + b;
        }
    }
}

// ---- host launcher --------------------------------------------------------
extern "C" void kernel_launch(void* const* d_in, const int* in_sizes, int n_in,
                              void* d_out, int out_size, void* d_ws, size_t ws_size,
                              hipStream_t stream) {
    const float* feat     = (const float*)d_in[0];
    const float* topk_val = (const float*)d_in[1];
    const int*   topk_idx = (const int*)  d_in[2];
    const int*   row      = (const int*)  d_in[3];
    const int*   col      = (const int*)  d_in[4];
    const float* degrees  = (const float*)d_in[5];
    const float* W_neigh  = (const float*)d_in[6];
    const float* W_self   = (const float*)d_in[7];
    const float* bias     = (const float*)d_in[8];
    float*       out      = (float*)d_out;

    const int n_edges = in_sizes[3];
    const int n_nodes = in_sizes[5];

    // workspace layout: indptr | packed W hi | packed W lo
    char* ws = (char*)d_ws;
    int*  indptr = (int*)ws;
    size_t ofs = (((size_t)(n_nodes + 1) * sizeof(int)) + 255) & ~(size_t)255;
    unsigned* bhi = (unsigned*)(ws + ofs);
    unsigned* blo = bhi + (size_t)N_STEPS * N_NTILES * 32 * 8;

    build_indptr_kernel<<<(n_edges + 255) / 256, 256, 0, stream>>>(
        row, indptr, n_edges, n_nodes);

    pack_W_kernel<<<(N_STEPS * N_NTILES * 32 + 255) / 256, 256, 0, stream>>>(
        W_self, W_neigh, bhi, blo);

    const int m_tiles = (n_nodes + 15) / 16;
    maxk_sage_fused_kernel<<<m_tiles, 256, 0, stream>>>(
        feat, topk_val, topk_idx, col, indptr, degrees,
        bhi, blo, bias, out, n_nodes);
}
